// MultiHeadAttention_66305705116361
// MI455X (gfx1250) — matrix operand order repaired and checked by reading, not compiled, not run on hardware
//
#include <hip/hip_runtime.h>
#include <math.h>

// ---------------------------------------------------------------------------
// MHA for MI455X (gfx1250): bf16 WMMA everywhere, flash-style fused attention
// with async global->LDS double-buffered K/V staging (ASYNCcnt path).
// B=2 S=4096 D=512 H=8 DK=64
// ---------------------------------------------------------------------------

typedef __bf16 bf16;
typedef __attribute__((ext_vector_type(16))) __bf16 v16bf;
typedef __attribute__((ext_vector_type(8)))  __bf16 v8bf;
typedef __attribute__((ext_vector_type(8)))  float  v8f;

static constexpr int B_  = 2;
static constexpr int S_  = 4096;
static constexpr int D_  = 512;
static constexpr int H_  = 8;
static constexpr int DK_ = 64;
static constexpr int NTOK = B_ * S_;     // 8192 token rows

#define DEV __device__ __forceinline__
#define NEG_INF (-__builtin_inff())

// --------------------------- WMMA fragment helpers -------------------------
// A-matrix 16x32 bf16 (ISA 7.12.2): lane m=lane&15; hi=lane>>4;
//   elems 0..7  -> k = i + 8*hi ;  elems 8..15 -> k = i + 8 + 8*hi
// B-matrix 32x16 bf16: lane n=lane&15; same k order as A (i + 8*hi / i + 8 + 8*hi).
// C/D 16x16 f32: lane n=lane&15; vgpr r -> row m = r + 8*hi.

DEV v16bf frag_cat(v8bf a, v8bf b) {
  return __builtin_shufflevector(a, b, 0,1,2,3,4,5,6,7,8,9,10,11,12,13,14,15);
}
DEV v16bf load_frag_a(const bf16* p, int hi) {
  v8bf c0 = *(const v8bf*)(p + hi * 8);
  v8bf c1 = *(const v8bf*)(p + 16 + hi * 8);
  return frag_cat(c0, c1);
}
DEV v16bf load_frag_b(const bf16* p, int hi) {
  v8bf c0 = *(const v8bf*)(p + hi * 8);
  v8bf c1 = *(const v8bf*)(p + 16 + hi * 8);
  return frag_cat(c0, c1);
}
DEV v8f wmma_bf16(v16bf a, v16bf b, v8f c) {
  return __builtin_amdgcn_wmma_f32_16x16x32_bf16(false, a, false, b, (short)0, c,
                                                 false, false);
}

// --------------------- async global -> LDS (ASYNCcnt) ----------------------
// 16B per lane per instruction. LDS dest address = low 32 bits of the generic
// shared pointer (LDS aperture maps addr[31:0] -> LDS offset).
DEV void async_cp16(const bf16* g, const bf16* l) {
  unsigned loff = (unsigned)(uintptr_t)l;
  unsigned long long ga = (unsigned long long)(uintptr_t)g;
  asm volatile("global_load_async_to_lds_b128 %0, %1, off"
               :: "v"(loff), "v"(ga) : "memory");
}
DEV void wait_async(int pending16) {   // pending16: 1 -> allow 16 in flight
  if (pending16)
    asm volatile("s_wait_asynccnt 0x10" ::: "memory");
  else
    asm volatile("s_wait_asynccnt 0x0" ::: "memory");
}

// Stage K tile: 32 rows x 128B (row stride D_ elems) -> LDS [32][64].
DEV void stage_K(const bf16* __restrict__ Kg, bf16* Kl, int lane) {
#pragma unroll
  for (int i = 0; i < 8; ++i) {
    int id = i * 32 + lane;
    int r = id >> 3, c = id & 7;
    async_cp16(Kg + (size_t)r * D_ + c * 8, Kl + r * 64 + c * 8);
  }
}
// Stage V tile: 64 rows x 64B (row stride S_ elems) -> LDS [64][32].
DEV void stage_V(const bf16* __restrict__ Vg, bf16* Vl, int lane) {
#pragma unroll
  for (int i = 0; i < 8; ++i) {
    int id = i * 32 + lane;
    int r = id >> 2, c = id & 3;
    async_cp16(Vg + (size_t)r * S_ + c * 8, Vl + r * 32 + c * 8);
  }
}

// ------------------------------ f32 -> bf16 --------------------------------
__global__ void __launch_bounds__(256)
cvt_f32_bf16_kernel(const float* __restrict__ in, bf16* __restrict__ out, int n) {
  int i = blockIdx.x * 256 + threadIdx.x;
  if (i < n) out[i] = (bf16)in[i];
}

// --------------------- shared 16x64 GEMM tile core -------------------------
// Register double-buffered: next k-step fragments are in flight while the
// current k-step's 4 WMMAs execute (no s_wait_loadcnt 0 per step).
DEV void gemm_core_16x64(const bf16* __restrict__ A, const bf16* __restrict__ W,
                         int K, int m0, int n0, int l16, int hi, v8f acc[4]) {
  const bf16* arow = A + (size_t)(m0 + l16) * K;
  const bf16* wrow = W + (size_t)(n0 + l16) * K;
  v16bf af = load_frag_a(arow, hi);
  v16bf b0 = load_frag_b(wrow, hi);
  v16bf b1 = load_frag_b(wrow + (size_t)16 * K, hi);
  v16bf b2 = load_frag_b(wrow + (size_t)32 * K, hi);
  v16bf b3 = load_frag_b(wrow + (size_t)48 * K, hi);
  for (int k0 = 32; k0 < K; k0 += 32) {
    __builtin_prefetch(arow + k0 + 256, 0, 3);
    v16bf afn = load_frag_a(arow + k0, hi);
    v16bf b0n = load_frag_b(wrow + k0, hi);
    v16bf b1n = load_frag_b(wrow + (size_t)16 * K + k0, hi);
    v16bf b2n = load_frag_b(wrow + (size_t)32 * K + k0, hi);
    v16bf b3n = load_frag_b(wrow + (size_t)48 * K + k0, hi);
    acc[0] = wmma_bf16(af, b0, acc[0]);
    acc[1] = wmma_bf16(af, b1, acc[1]);
    acc[2] = wmma_bf16(af, b2, acc[2]);
    acc[3] = wmma_bf16(af, b3, acc[3]);
    af = afn; b0 = b0n; b1 = b1n; b2 = b2n; b3 = b3n;
  }
  acc[0] = wmma_bf16(af, b0, acc[0]);
  acc[1] = wmma_bf16(af, b1, acc[1]);
  acc[2] = wmma_bf16(af, b2, acc[2]);
  acc[3] = wmma_bf16(af, b3, acc[3]);
}

DEV bool gemm_task(int* m0, int* n0, int* l16, int* hi) {
  int lane = threadIdx.x & 31;
  int wid  = threadIdx.x >> 5;
  int task = blockIdx.x * 4 + wid;          // (NTOK/16)*(D/64) = 4096 tasks
  *m0 = (task >> 3) * 16;
  *n0 = (task & 7) * 64;
  *l16 = lane & 15;
  *hi  = lane >> 4;
  return *m0 < NTOK;
}

__global__ void __launch_bounds__(128)
gemm_xWT_bf16_kernel(const bf16* __restrict__ A, const bf16* __restrict__ W,
                     bf16* __restrict__ C) {
  int m0, n0, l16, hi;
  if (!gemm_task(&m0, &n0, &l16, &hi)) return;
  v8f acc[4] = {v8f{}, v8f{}, v8f{}, v8f{}};
  gemm_core_16x64(A, W, D_, m0, n0, l16, hi, acc);
#pragma unroll
  for (int t = 0; t < 4; ++t) {
    int col = n0 + t * 16 + l16;
#pragma unroll
    for (int r = 0; r < 8; ++r)
      C[(size_t)(m0 + r + 8 * hi) * D_ + col] = (bf16)acc[t][r];
  }
}

__global__ void __launch_bounds__(128)
gemm_xWT_vt_kernel(const bf16* __restrict__ A, const bf16* __restrict__ W,
                   bf16* __restrict__ Vt) {
  int m0, n0, l16, hi;
  if (!gemm_task(&m0, &n0, &l16, &hi)) return;
  v8f acc[4] = {v8f{}, v8f{}, v8f{}, v8f{}};
  gemm_core_16x64(A, W, D_, m0, n0, l16, hi, acc);
  int b  = m0 / S_;
  int s0 = m0 % S_;
#pragma unroll
  for (int t = 0; t < 4; ++t) {
    int n = n0 + t * 16 + l16;               // n = h*64 + dk
    v8bf pk;
#pragma unroll
    for (int r = 0; r < 8; ++r) pk[r] = (bf16)acc[t][r];
    *(v8bf*)(Vt + (size_t)(b * 512 + n) * S_ + s0 + 8 * hi) = pk;
  }
}

__global__ void __launch_bounds__(128)
gemm_xWT_f32_kernel(const bf16* __restrict__ A, const bf16* __restrict__ W,
                    float* __restrict__ C) {
  int m0, n0, l16, hi;
  if (!gemm_task(&m0, &n0, &l16, &hi)) return;
  v8f acc[4] = {v8f{}, v8f{}, v8f{}, v8f{}};
  gemm_core_16x64(A, W, D_, m0, n0, l16, hi, acc);
#pragma unroll
  for (int t = 0; t < 4; ++t) {
    int col = n0 + t * 16 + l16;
#pragma unroll
    for (int r = 0; r < 8; ++r)
      C[(size_t)(m0 + r + 8 * hi) * D_ + col] = acc[t][r];
  }
}

// ------------------------------ flash attention ----------------------------
// One wave per (b, h, 16-query tile); K/V tiles double-buffered into LDS via
// async DMA; WMMA fragments read from LDS (ds_load_b128).
static constexpr int KV_ELEMS = 2048;              // 32x64 or 64x32 bf16
static constexpr int WAVE_LDS = 4 * KV_ELEMS + 512;  // K[2] V[2] P

__global__ void __launch_bounds__(128)
flash_attn_kernel(const bf16* __restrict__ Qp, const bf16* __restrict__ Kp,
                  const bf16* __restrict__ Vt, const int* __restrict__ mask,
                  bf16* __restrict__ AO) {
  __shared__ __align__(16) bf16 smem[4][WAVE_LDS];
  const int lane = threadIdx.x & 31;
  const int wid  = threadIdx.x >> 5;
  const int l16  = lane & 15;
  const int hi   = lane >> 4;
  int task = blockIdx.x * 4 + wid;           // B*H*(S/16) = 4096 tasks
  int qt = task & 255;
  int h  = (task >> 8) & 7;
  int b  = task >> 11;
  int q0 = qt * 16;

  bf16* sw   = &smem[wid][0];
  bf16* Kb[2] = { sw,              sw + KV_ELEMS };
  bf16* Vb[2] = { sw + 2*KV_ELEMS, sw + 3*KV_ELEMS };
  bf16* Pl    = sw + 4*KV_ELEMS;

  const bf16* Kg = Kp + (size_t)(b * S_) * D_ + h * DK_;          // + kk*D_
  const bf16* Vg = Vt + (size_t)((b * H_ + h) * DK_) * S_;        // + kk

  const bf16* qrow = Qp + (size_t)(b * S_ + q0 + l16) * D_ + h * DK_;
  v16bf qf0 = load_frag_a(qrow, hi);
  v16bf qf1 = load_frag_a(qrow + 32, hi);

  v8f O0 = {}, O1 = {}, O2 = {}, O3 = {};
  float Mst[8], Lst[8];
#pragma unroll
  for (int r = 0; r < 8; ++r) { Mst[r] = NEG_INF; Lst[r] = 0.f; }

  const float scale = 0.125f;                // 1/sqrt(64)
  const int*  mrow  = mask + b * S_;

  // prime buffer 0
  stage_K(Kg, Kb[0], lane);
  stage_V(Vg, Vb[0], lane);
  int cur = 0;

  for (int kk = 0; kk < S_; kk += 32) {
    if (kk + 32 < S_) {                      // stage next tile, keep 16 in flight
      stage_K(Kg + (size_t)(kk + 32) * D_, Kb[cur ^ 1], lane);
      stage_V(Vg + (kk + 32),              Vb[cur ^ 1], lane);
      wait_async(1);
    } else {
      wait_async(0);
    }
    const bf16* Kl = Kb[cur];
    const bf16* Vl = Vb[cur];

    // ---- scores from LDS fragments -------------------------------------
    v8f s0 = {}, s1 = {};
    s0 = wmma_bf16(qf0, load_frag_b(Kl + l16 * 64,            hi), s0);
    s0 = wmma_bf16(qf1, load_frag_b(Kl + l16 * 64 + 32,       hi), s0);
    s1 = wmma_bf16(qf0, load_frag_b(Kl + (16 + l16) * 64,     hi), s1);
    s1 = wmma_bf16(qf1, load_frag_b(Kl + (16 + l16) * 64 + 32, hi), s1);

    int mk0 = mrow[kk + l16];
    int mk1 = mrow[kk + 16 + l16];

    float p0[8], p1[8];
#pragma unroll
    for (int r = 0; r < 8; ++r) {
      float a0 = mk0 ? s0[r] * scale : NEG_INF;
      float a1 = mk1 ? s1[r] * scale : NEG_INF;
      float mx = fmaxf(a0, a1);
      mx = fmaxf(mx, __shfl_xor(mx, 1));
      mx = fmaxf(mx, __shfl_xor(mx, 2));
      mx = fmaxf(mx, __shfl_xor(mx, 4));
      mx = fmaxf(mx, __shfl_xor(mx, 8));
      float Mnew = fmaxf(Mst[r], mx);
      float c, e0, e1;
      if (Mnew == NEG_INF) { c = 1.f; e0 = 0.f; e1 = 0.f; }
      else {
        c  = __expf(Mst[r] - Mnew);
        e0 = __expf(a0 - Mnew);
        e1 = __expf(a1 - Mnew);
      }
      float sm = e0 + e1;
      sm += __shfl_xor(sm, 1);
      sm += __shfl_xor(sm, 2);
      sm += __shfl_xor(sm, 4);
      sm += __shfl_xor(sm, 8);
      Lst[r] = Lst[r] * c + sm;
      Mst[r] = Mnew;
      O0[r] *= c; O1[r] *= c; O2[r] *= c; O3[r] *= c;
      p0[r] = e0; p1[r] = e1;
    }

    // ---- P (D-layout) -> LDS -> A-fragment -----------------------------
#pragma unroll
    for (int r = 0; r < 8; ++r) {
      Pl[(r + 8 * hi) * 32 + l16]      = (bf16)p0[r];
      Pl[(r + 8 * hi) * 32 + l16 + 16] = (bf16)p1[r];
    }
    __syncthreads();
    v16bf pf = load_frag_a(Pl + l16 * 32, hi);

    // ---- O += P * V from LDS -------------------------------------------
    O0 = wmma_bf16(pf, load_frag_b(Vl + (     l16) * 32, hi), O0);
    O1 = wmma_bf16(pf, load_frag_b(Vl + (16 + l16) * 32, hi), O1);
    O2 = wmma_bf16(pf, load_frag_b(Vl + (32 + l16) * 32, hi), O2);
    O3 = wmma_bf16(pf, load_frag_b(Vl + (48 + l16) * 32, hi), O3);
    __syncthreads();
    cur ^= 1;
  }

  // ---- epilogue: O / l (l==0 -> 0, matches nan_to_num) ------------------
#pragma unroll
  for (int r = 0; r < 8; ++r) {
    float inv = Lst[r] > 0.f ? 1.f / Lst[r] : 0.f;
    bf16* orow = AO + (size_t)(b * S_ + q0 + r + 8 * hi) * D_ + h * DK_ + l16;
    orow[0]  = (bf16)(O0[r] * inv);
    orow[16] = (bf16)(O1[r] * inv);
    orow[32] = (bf16)(O2[r] * inv);
    orow[48] = (bf16)(O3[r] * inv);
  }
}

// ------------------------------- launch ------------------------------------
extern "C" void kernel_launch(void* const* d_in, const int* in_sizes, int n_in,
                              void* d_out, int out_size, void* d_ws, size_t ws_size,
                              hipStream_t stream) {
  (void)in_sizes; (void)n_in; (void)out_size; (void)ws_size;
  const float* query = (const float*)d_in[0];
  const float* key   = (const float*)d_in[1];
  const float* value = (const float*)d_in[2];
  const int*   mask  = (const int*)d_in[3];
  const float* Wq = (const float*)d_in[4];
  const float* Wk = (const float*)d_in[5];
  const float* Wv = (const float*)d_in[6];
  const float* Wo = (const float*)d_in[7];

  char* ws = (char*)d_ws;
  size_t off = 0;
  auto alloc = [&](size_t bytes) -> char* {
    char* p = ws + off;
    off += (bytes + 255) & ~(size_t)255;
    return p;
  };
  const size_t actB = (size_t)NTOK * D_ * sizeof(bf16);   // 8 MB
  const size_t wB   = (size_t)D_ * D_ * sizeof(bf16);     // 512 KB
  bf16* qb  = (bf16*)alloc(actB);
  bf16* kb  = (bf16*)alloc(actB);
  bf16* vb  = (bf16*)alloc(actB);
  bf16* wqb = (bf16*)alloc(wB);
  bf16* wkb = (bf16*)alloc(wB);
  bf16* wvb = (bf16*)alloc(wB);
  bf16* wob = (bf16*)alloc(wB);
  bf16* Qp  = (bf16*)alloc(actB);            // [NTOK, 512]
  bf16* Kp  = (bf16*)alloc(actB);            // [NTOK, 512]
  bf16* Vt  = (bf16*)alloc(actB);            // [B, H, DK, S]
  bf16* AO  = (bf16*)alloc(actB);            // attention output [NTOK, 512]

  const int nAct = NTOK * D_;
  const int nW   = D_ * D_;
  cvt_f32_bf16_kernel<<<(nAct + 255) / 256, 256, 0, stream>>>(query, qb, nAct);
  cvt_f32_bf16_kernel<<<(nAct + 255) / 256, 256, 0, stream>>>(key,   kb, nAct);
  cvt_f32_bf16_kernel<<<(nAct + 255) / 256, 256, 0, stream>>>(value, vb, nAct);
  cvt_f32_bf16_kernel<<<(nW + 255) / 256, 256, 0, stream>>>(Wq, wqb, nW);
  cvt_f32_bf16_kernel<<<(nW + 255) / 256, 256, 0, stream>>>(Wk, wkb, nW);
  cvt_f32_bf16_kernel<<<(nW + 255) / 256, 256, 0, stream>>>(Wv, wvb, nW);
  cvt_f32_bf16_kernel<<<(nW + 255) / 256, 256, 0, stream>>>(Wo, wob, nW);

  const int gemmBlocks = (NTOK / 16) * (D_ / 64) / 4;      // 1024
  gemm_xWT_bf16_kernel<<<gemmBlocks, 128, 0, stream>>>(qb, wqb, Qp);
  gemm_xWT_bf16_kernel<<<gemmBlocks, 128, 0, stream>>>(kb, wkb, Kp);
  gemm_xWT_vt_kernel  <<<gemmBlocks, 128, 0, stream>>>(vb, wvb, Vt);

  const int flashBlocks = (B_ * H_ * (S_ / 16)) / 4;       // 1024
  flash_attn_kernel<<<flashBlocks, 128, 0, stream>>>(Qp, Kp, Vt, mask, AO);

  gemm_xWT_f32_kernel<<<gemmBlocks, 128, 0, stream>>>(AO, wob, (float*)d_out);
}
